// GMARAFT_Denoiser_9706626089136
// MI455X (gfx1250) — matrix operand order repaired and checked by reading, not compiled, not run on hardware
//
#include <hip/hip_runtime.h>
#include <math.h>

typedef __attribute__((ext_vector_type(16))) _Float16 v16h;
typedef __attribute__((ext_vector_type(8)))  _Float16 v8h;
typedef __attribute__((ext_vector_type(8)))  float    v8f;

#define LDT 40   // padded LDS row stride (halves): 80B, 16B-aligned, bank-spread

__device__ __forceinline__ float actf(float v, int act) {
    if (act == 1) return v > 0.f ? v : 0.f;
    if (act == 2) return 1.f / (1.f + expf(-v));
    if (act == 3) return tanhf(v);
    return v;
}

// ---------------------------------------------------------------------------
// Implicit-GEMM convolution. 4 waves / block, 64(M) x 16(N) tile, K chunked
// by 32 for v_wmma_f32_16x16x32_f16. B (im2col gather) staged once per chunk
// and shared by all 4 waves; LDS double-buffered, last chunk peeled.
// Staging keeps loads unconditional (clamped index + mask-multiply).
//   y[n, c0+m, oy, ox] = act( (conv + bias[m]) * alpha )
// x: [Nb, Cin, Hi, Wi]   w: [Cout, Cin, KH, KW]   y: [Nb, Ctot, Ho, Wo]
// ---------------------------------------------------------------------------
__global__ __launch_bounds__(128)
void k_wmma_conv(const float* __restrict__ x, const float* __restrict__ w,
                 const float* __restrict__ bias, float* __restrict__ y,
                 int Nb, int Cin, int Hi, int Wi, int Cout, int Ho, int Wo,
                 int KH, int KW, int stride, int pt, int pl,
                 int c0, int Ctot, float alpha, int act) {
    __shared__ __align__(16) _Float16 lA[2][64 * LDT];
    __shared__ __align__(16) _Float16 lB[2][16 * LDT];

    const int t  = threadIdx.x;
    const int wv = t >> 5;              // wave id 0..3
    const int l  = t & 31;
    const int lm = l & 15;
    const int hi = l >> 4;
    const int ks = hi * 16;

    const int P     = Ho * Wo;
    const int Ncols = Nb * P;
    const int colBase = blockIdx.x * 16;
    const int mBase   = blockIdx.y * 64;
    const int KHW = KH * KW;
    const int K   = Cin * KHW;

    // --- staging roles (whole 128-thread block) ---
    const int sArow = t >> 1;           // A: row 0..63, k-span 16
    const int sAk   = (t & 1) * 16;
    const int sAm   = mBase + sArow;
    const bool sAmv = sAm < Cout;
    const int sAmc  = sAmv ? sAm : (Cout - 1);

    const int sBc   = t >> 3;           // B: col 0..15, k-span 4
    const int sBk   = (t & 7) * 4;
    const int sBcol = colBase + sBc;
    const bool sBcv = sBcol < Ncols;
    const int sBcolc = sBcv ? sBcol : (Ncols - 1);
    const int bn  = sBcolc / P;
    const int bp  = sBcolc - bn * P;
    const int boy = bp / Wo;
    const int box = bp - boy * Wo;
    const int iy0 = boy * stride - pt;
    const int ix0 = box * stride - pl;
    const float bmask = sBcv ? 1.f : 0.f;

    auto stage = [&](int kb, _Float16* A, _Float16* B) {
        // ---- A (weights): K-contiguous ----
        const int abase = kb + sAk;
        if (sAmv && (abase + 15 < K)) {
            const float* src = w + (size_t)sAmc * K + abase;
            #pragma unroll
            for (int j = 0; j < 16; ++j)
                A[sArow * LDT + sAk + j] = (_Float16)src[j];
        } else {
            #pragma unroll
            for (int j = 0; j < 16; ++j) {
                int k  = abase + j;
                int kc = k < K ? k : (K - 1);
                float v = w[(size_t)sAmc * K + kc];
                v *= (sAmv && k < K) ? 1.f : 0.f;
                A[sArow * LDT + sAk + j] = (_Float16)v;
            }
        }
        // ---- B (im2col gather): incremental (ci,ky,kx) decode ----
        int k0 = kb + sBk;
        int ci  = k0 / KHW;
        int rem = k0 - ci * KHW;
        int ky  = rem / KW;
        int kx  = rem - ky * KW;
        #pragma unroll
        for (int j = 0; j < 4; ++j) {
            int k  = k0 + j;
            int iy = iy0 + ky;
            int ix = ix0 + kx;
            float msk = ((k < K) && iy >= 0 && iy < Hi && ix >= 0 && ix < Wi) ? bmask : 0.f;
            int iyc = iy < 0 ? 0 : (iy >= Hi ? Hi - 1 : iy);
            int ixc = ix < 0 ? 0 : (ix >= Wi ? Wi - 1 : ix);
            int cic = ci < Cin ? ci : (Cin - 1);
            float v = x[(((size_t)bn * Cin + cic) * Hi + iyc) * Wi + ixc];
            B[sBc * LDT + sBk + j] = (_Float16)(v * msk);
            ++kx;
            if (kx == KW) { kx = 0; ++ky; if (ky == KH) { ky = 0; ++ci; } }
        }
    };

    v8f acc = {0.f, 0.f, 0.f, 0.f, 0.f, 0.f, 0.f, 0.f};
    const int nChunks = (K + 31) >> 5;

    stage(0, lA[0], lB[0]);
    __syncthreads();
    int cur = 0;
    for (int c = 0; c + 1 < nChunks; ++c) {
        const _Float16* A = lA[cur];
        const _Float16* B = lB[cur];
        v8h a0 = *(const v8h*)(A + (wv * 16 + lm) * LDT + ks);
        v8h a1 = *(const v8h*)(A + (wv * 16 + lm) * LDT + ks + 8);
        v8h b0 = *(const v8h*)(B + lm * LDT + ks);
        v8h b1 = *(const v8h*)(B + lm * LDT + ks + 8);
        v16h av, bv;
        #pragma unroll
        for (int i = 0; i < 8; ++i) {
            av[i] = a0[i]; av[i + 8] = a1[i];
            bv[i] = b0[i]; bv[i + 8] = b1[i];
        }
        stage((c + 1) * 32, lA[cur ^ 1], lB[cur ^ 1]);
        acc = __builtin_amdgcn_wmma_f32_16x16x32_f16(false, av, false, bv,
                                                     (short)0, acc, false, false);
        __syncthreads();
        cur ^= 1;
    }
    {   // last chunk (peeled)
        const _Float16* A = lA[cur];
        const _Float16* B = lB[cur];
        v8h a0 = *(const v8h*)(A + (wv * 16 + lm) * LDT + ks);
        v8h a1 = *(const v8h*)(A + (wv * 16 + lm) * LDT + ks + 8);
        v8h b0 = *(const v8h*)(B + lm * LDT + ks);
        v8h b1 = *(const v8h*)(B + lm * LDT + ks + 8);
        v16h av, bv;
        #pragma unroll
        for (int i = 0; i < 8; ++i) {
            av[i] = a0[i]; av[i + 8] = a1[i];
            bv[i] = b0[i]; bv[i + 8] = b1[i];
        }
        acc = __builtin_amdgcn_wmma_f32_16x16x32_f16(false, av, false, bv,
                                                     (short)0, acc, false, false);
    }

    const int col = colBase + lm;
    if (col < Ncols) {
        int n = col / P;
        int p = col - n * P;
        #pragma unroll
        for (int r = 0; r < 8; ++r) {
            int mrow = mBase + wv * 16 + hi * 8 + r;
            if (mrow < Cout) {
                float v = (acc[r] + bias[mrow]) * alpha;
                v = actf(v, act);
                y[(((size_t)n * Ctot + c0 + mrow) * P) + p] = v;
            }
        }
    }
}

// ---------------------------------------------------------------------------
// Batched WMMA GEMM:  out[z, m, col] = alpha * sum_k A(z,m,k) * B(z,k,col)
// tA: A stored [z,K,M]; tB: B stored [z,Ncols,K]
// ---------------------------------------------------------------------------
__global__ __launch_bounds__(128)
void k_wmma_gemm(const float* __restrict__ a, const float* __restrict__ b,
                 float* __restrict__ out, int M, int Ncols, int K,
                 int tA, int tB, float alpha) {
    __shared__ __align__(16) _Float16 lA[2][64 * LDT];
    __shared__ __align__(16) _Float16 lB[2][16 * LDT];

    const int t  = threadIdx.x;
    const int wv = t >> 5;
    const int l  = t & 31;
    const int lm = l & 15;
    const int hi = l >> 4;
    const int ks = hi * 16;

    a   += (size_t)blockIdx.z * M * K;
    b   += (size_t)blockIdx.z * Ncols * K;
    out += (size_t)blockIdx.z * M * Ncols;

    const int colBase = blockIdx.x * 16;
    const int mBase   = blockIdx.y * 64;

    const int sArow = t >> 1;
    const int sAk   = (t & 1) * 16;
    const int sAm   = mBase + sArow;
    const bool sAmv = sAm < M;
    const int sAmc  = sAmv ? sAm : (M - 1);

    const int sBc   = t >> 3;
    const int sBk   = (t & 7) * 4;
    const int sBcol = colBase + sBc;
    const bool sBcv = sBcol < Ncols;
    const int sBcolc = sBcv ? sBcol : (Ncols - 1);

    auto stage = [&](int kb, _Float16* A, _Float16* B) {
        const int abase = kb + sAk;
        if (sAmv && (abase + 15 < K)) {
            #pragma unroll
            for (int j = 0; j < 16; ++j) {
                int k = abase + j;
                float v = tA ? a[(size_t)k * M + sAmc] : a[(size_t)sAmc * K + k];
                A[sArow * LDT + sAk + j] = (_Float16)v;
            }
        } else {
            #pragma unroll
            for (int j = 0; j < 16; ++j) {
                int k  = abase + j;
                int kc = k < K ? k : (K - 1);
                float v = tA ? a[(size_t)kc * M + sAmc] : a[(size_t)sAmc * K + kc];
                v *= (sAmv && k < K) ? 1.f : 0.f;
                A[sArow * LDT + sAk + j] = (_Float16)v;
            }
        }
        const int bbase = kb + sBk;
        if (sBcv && (bbase + 3 < K)) {
            #pragma unroll
            for (int j = 0; j < 4; ++j) {
                int k = bbase + j;
                float v = tB ? b[(size_t)sBcolc * K + k] : b[(size_t)k * Ncols + sBcolc];
                B[sBc * LDT + sBk + j] = (_Float16)v;
            }
        } else {
            #pragma unroll
            for (int j = 0; j < 4; ++j) {
                int k  = bbase + j;
                int kc = k < K ? k : (K - 1);
                float v = tB ? b[(size_t)sBcolc * K + kc] : b[(size_t)kc * Ncols + sBcolc];
                v *= (sBcv && k < K) ? 1.f : 0.f;
                B[sBc * LDT + sBk + j] = (_Float16)v;
            }
        }
    };

    v8f acc = {0.f, 0.f, 0.f, 0.f, 0.f, 0.f, 0.f, 0.f};
    const int nChunks = (K + 31) >> 5;

    stage(0, lA[0], lB[0]);
    __syncthreads();
    int cur = 0;
    for (int c = 0; c + 1 < nChunks; ++c) {
        const _Float16* A = lA[cur];
        const _Float16* B = lB[cur];
        v8h a0 = *(const v8h*)(A + (wv * 16 + lm) * LDT + ks);
        v8h a1 = *(const v8h*)(A + (wv * 16 + lm) * LDT + ks + 8);
        v8h b0 = *(const v8h*)(B + lm * LDT + ks);
        v8h b1 = *(const v8h*)(B + lm * LDT + ks + 8);
        v16h av, bv;
        #pragma unroll
        for (int i = 0; i < 8; ++i) {
            av[i] = a0[i]; av[i + 8] = a1[i];
            bv[i] = b0[i]; bv[i + 8] = b1[i];
        }
        stage((c + 1) * 32, lA[cur ^ 1], lB[cur ^ 1]);
        acc = __builtin_amdgcn_wmma_f32_16x16x32_f16(false, av, false, bv,
                                                     (short)0, acc, false, false);
        __syncthreads();
        cur ^= 1;
    }
    {   // last chunk (peeled)
        const _Float16* A = lA[cur];
        const _Float16* B = lB[cur];
        v8h a0 = *(const v8h*)(A + (wv * 16 + lm) * LDT + ks);
        v8h a1 = *(const v8h*)(A + (wv * 16 + lm) * LDT + ks + 8);
        v8h b0 = *(const v8h*)(B + lm * LDT + ks);
        v8h b1 = *(const v8h*)(B + lm * LDT + ks + 8);
        v16h av, bv;
        #pragma unroll
        for (int i = 0; i < 8; ++i) {
            av[i] = a0[i]; av[i + 8] = a1[i];
            bv[i] = b0[i]; bv[i + 8] = b1[i];
        }
        acc = __builtin_amdgcn_wmma_f32_16x16x32_f16(false, av, false, bv,
                                                     (short)0, acc, false, false);
    }

    const int col = colBase + lm;
    if (col < Ncols) {
        #pragma unroll
        for (int r = 0; r < 8; ++r) {
            int mrow = mBase + wv * 16 + hi * 8 + r;
            if (mrow < M) out[(size_t)mrow * Ncols + col] = acc[r] * alpha;
        }
    }
}

// --------------------------- norm / softmax --------------------------------
__global__ void k_inorm(float* x, int P, int relu) {
    __shared__ float s1[256];
    __shared__ float s2[256];
    float* p = x + (size_t)blockIdx.x * P;
    float s = 0.f, q = 0.f;
    for (int i = threadIdx.x; i < P; i += 256) { float v = p[i]; s += v; q += v * v; }
    s1[threadIdx.x] = s; s2[threadIdx.x] = q;
    __syncthreads();
    for (int o = 128; o > 0; o >>= 1) {
        if ((int)threadIdx.x < o) { s1[threadIdx.x] += s1[threadIdx.x + o]; s2[threadIdx.x] += s2[threadIdx.x + o]; }
        __syncthreads();
    }
    float mean = s1[0] / P;
    float var  = s2[0] / P - mean * mean;
    float inv  = rsqrtf(var + 1e-5f);
    for (int i = threadIdx.x; i < P; i += 256) {
        float v = (p[i] - mean) * inv;
        if (relu) v = fmaxf(v, 0.f);
        p[i] = v;
    }
}

__global__ void k_bnorm(float* x, int N, int C, int P, int relu) {
    __shared__ float s1[256];
    __shared__ float s2[256];
    const int c = blockIdx.x;
    const int total = N * P;
    float s = 0.f, q = 0.f;
    for (int i = threadIdx.x; i < total; i += 256) {
        int n = i / P, pp = i - n * P;
        float v = x[(((size_t)n * C + c) * P) + pp];
        s += v; q += v * v;
    }
    s1[threadIdx.x] = s; s2[threadIdx.x] = q;
    __syncthreads();
    for (int o = 128; o > 0; o >>= 1) {
        if ((int)threadIdx.x < o) { s1[threadIdx.x] += s1[threadIdx.x + o]; s2[threadIdx.x] += s2[threadIdx.x + o]; }
        __syncthreads();
    }
    float mean = s1[0] / total;
    float var  = s2[0] / total - mean * mean;
    float inv  = rsqrtf(var + 1e-5f);
    for (int i = threadIdx.x; i < total; i += 256) {
        int n = i / P, pp = i - n * P;
        size_t idx = (((size_t)n * C + c) * P) + pp;
        float v = (x[idx] - mean) * inv;
        if (relu) v = fmaxf(v, 0.f);
        x[idx] = v;
    }
}

__global__ void k_softmax_rows(float* x, int cols) {
    __shared__ float sh[256];
    float* row = x + (size_t)blockIdx.x * cols;
    float mx = -3.0e38f;
    for (int i = threadIdx.x; i < cols; i += 256) mx = fmaxf(mx, row[i]);
    sh[threadIdx.x] = mx; __syncthreads();
    for (int o = 128; o > 0; o >>= 1) {
        if ((int)threadIdx.x < o) sh[threadIdx.x] = fmaxf(sh[threadIdx.x], sh[threadIdx.x + o]);
        __syncthreads();
    }
    mx = sh[0]; __syncthreads();
    float s = 0.f;
    for (int i = threadIdx.x; i < cols; i += 256) { float e = expf(row[i] - mx); row[i] = e; s += e; }
    sh[threadIdx.x] = s; __syncthreads();
    for (int o = 128; o > 0; o >>= 1) {
        if ((int)threadIdx.x < o) sh[threadIdx.x] += sh[threadIdx.x + o];
        __syncthreads();
    }
    float inv = 1.f / sh[0];
    for (int i = threadIdx.x; i < cols; i += 256) row[i] *= inv;
}

// ------------------------------ elementwise --------------------------------
__global__ void k_scale(const float* __restrict__ x, float* __restrict__ y, int n) {
    int t = blockIdx.x * blockDim.x + threadIdx.x;
    if (t < n) y[t] = x[t] * (2.f / 255.f) - 1.f;
}

__global__ void k_copy_ch(const float* __restrict__ src, float* __restrict__ dst,
                          int Nb, int Cs, int Cd, int off, int P) {
    int t = blockIdx.x * blockDim.x + threadIdx.x;
    int total = Nb * Cs * P;
    if (t >= total) return;
    int n = t / (Cs * P);
    int r = t - n * (Cs * P);
    dst[((size_t)n * Cd + off) * P + r] = src[t];
}

__global__ void k_mul_into(const float* __restrict__ a, const float* __restrict__ b,
                           float* __restrict__ dst, int Nb, int Cs, int Cd, int off, int P) {
    int t = blockIdx.x * blockDim.x + threadIdx.x;
    int total = Nb * Cs * P;
    if (t >= total) return;
    int n = t / (Cs * P);
    int r = t - n * (Cs * P);
    dst[((size_t)n * Cd + off) * P + r] = a[t] * b[t];
}

__global__ void k_gru_update(float* __restrict__ h, const float* __restrict__ z,
                             const float* __restrict__ q, int n) {
    int t = blockIdx.x * blockDim.x + threadIdx.x;
    if (t < n) h[t] = (1.f - z[t]) * h[t] + z[t] * q[t];
}

__global__ void k_axpy_gamma(float* __restrict__ dst, const float* __restrict__ m,
                             const float* __restrict__ agg, const float* __restrict__ gamma, int n) {
    int t = blockIdx.x * blockDim.x + threadIdx.x;
    if (t < n) dst[t] = m[t] + gamma[0] * agg[t];
}

__global__ void k_resadd(float* __restrict__ cu, const float* __restrict__ a,
                         const float* __restrict__ b, int n) {
    int t = blockIdx.x * blockDim.x + threadIdx.x;
    if (t < n) cu[t] = a[t] + b[t];
}

__global__ void k_add_inplace(float* __restrict__ a, const float* __restrict__ b, int n) {
    int t = blockIdx.x * blockDim.x + threadIdx.x;
    if (t < n) a[t] += b[t];
}

// ------------------------------ reductions ---------------------------------
__global__ void k_minmax_part(const float* __restrict__ x, int n,
                              float* __restrict__ pmin, float* __restrict__ pmax) {
    __shared__ float smn[256];
    __shared__ float smx[256];
    float mn = 3.0e38f, mx = -3.0e38f;
    for (int i = blockIdx.x * 256 + threadIdx.x; i < n; i += gridDim.x * 256) {
        float v = x[i]; mn = fminf(mn, v); mx = fmaxf(mx, v);
    }
    smn[threadIdx.x] = mn; smx[threadIdx.x] = mx;
    __syncthreads();
    for (int o = 128; o > 0; o >>= 1) {
        if ((int)threadIdx.x < o) {
            smn[threadIdx.x] = fminf(smn[threadIdx.x], smn[threadIdx.x + o]);
            smx[threadIdx.x] = fmaxf(smx[threadIdx.x], smx[threadIdx.x + o]);
        }
        __syncthreads();
    }
    if (threadIdx.x == 0) { pmin[blockIdx.x] = smn[0]; pmax[blockIdx.x] = smx[0]; }
}

__global__ void k_minmax_fin(const float* __restrict__ pmin, const float* __restrict__ pmax,
                             float* __restrict__ mm, int nb) {
    __shared__ float smn[256];
    __shared__ float smx[256];
    float mn = 3.0e38f, mx = -3.0e38f;
    for (int i = threadIdx.x; i < nb; i += 256) { mn = fminf(mn, pmin[i]); mx = fmaxf(mx, pmax[i]); }
    smn[threadIdx.x] = mn; smx[threadIdx.x] = mx;
    __syncthreads();
    for (int o = 128; o > 0; o >>= 1) {
        if ((int)threadIdx.x < o) {
            smn[threadIdx.x] = fminf(smn[threadIdx.x], smn[threadIdx.x + o]);
            smx[threadIdx.x] = fmaxf(smx[threadIdx.x], smx[threadIdx.x + o]);
        }
        __syncthreads();
    }
    if (threadIdx.x == 0) { mm[0] = smn[0]; mm[1] = smx[0]; }
}

__global__ void k_renorm(float* __restrict__ cu, const float* __restrict__ mm,
                         float* __restrict__ out, int n) {
    int t = blockIdx.x * blockDim.x + threadIdx.x;
    if (t >= n) return;
    float mn = mm[0], mx = mm[1];
    float v = 2.f * ((cu[t] - mn) / (mx - mn)) - 1.f;
    cu[t] = v;
    out[t] = v;
}

// ------------------------- RAFT-specific kernels ---------------------------
__global__ void k_coords_init(float* __restrict__ c1, int Nb, int Hc, int Wc) {
    int t = blockIdx.x * blockDim.x + threadIdx.x;
    int P = Hc * Wc;
    if (t >= Nb * P) return;
    int n = t / P, p = t - n * P;
    c1[((size_t)n * 2 + 0) * P + p] = (float)(p % Wc);
    c1[((size_t)n * 2 + 1) * P + p] = (float)(p / Wc);
}

__global__ void k_flow(const float* __restrict__ c1, float* __restrict__ flow,
                       float* __restrict__ motion, int Nb, int Hc, int Wc) {
    int t = blockIdx.x * blockDim.x + threadIdx.x;
    int P = Hc * Wc;
    if (t >= Nb * P) return;
    int n = t / P, p = t - n * P;
    float fx = c1[((size_t)n * 2 + 0) * P + p] - (float)(p % Wc);
    float fy = c1[((size_t)n * 2 + 1) * P + p] - (float)(p / Wc);
    flow[((size_t)n * 2 + 0) * P + p] = fx;
    flow[((size_t)n * 2 + 1) * P + p] = fy;
    motion[((size_t)n * 128 + 126) * P + p] = fx;
    motion[((size_t)n * 128 + 127) * P + p] = fy;
}

__global__ void k_avgpool(const float* __restrict__ src, float* __restrict__ dst,
                          int M, int Ho, int Wo) {
    int t = blockIdx.x * blockDim.x + threadIdx.x;
    int total = M * Ho * Wo;
    if (t >= total) return;
    int m = t / (Ho * Wo);
    int r = t - m * (Ho * Wo);
    int h = r / Wo, w = r - h * Wo;
    int Wi = Wo * 2;
    const float* s = src + (size_t)m * (Ho * 2) * Wi;
    dst[t] = 0.25f * (s[(2 * h) * Wi + 2 * w] + s[(2 * h) * Wi + 2 * w + 1] +
                      s[(2 * h + 1) * Wi + 2 * w] + s[(2 * h + 1) * Wi + 2 * w + 1]);
}

__device__ __forceinline__ float bilin(const float* img, int H, int W, float x, float y) {
    float x0 = floorf(x), y0 = floorf(y);
    float wx = x - x0, wy = y - y0;
    int xi = (int)x0, yi = (int)y0;
    float v00 = 0.f, v10 = 0.f, v01 = 0.f, v11 = 0.f;
    if (xi >= 0 && xi < W && yi >= 0 && yi < H)             v00 = img[yi * W + xi];
    if (xi + 1 >= 0 && xi + 1 < W && yi >= 0 && yi < H)     v10 = img[yi * W + xi + 1];
    if (xi >= 0 && xi < W && yi + 1 >= 0 && yi + 1 < H)     v01 = img[(yi + 1) * W + xi];
    if (xi + 1 >= 0 && xi + 1 < W && yi + 1 >= 0 && yi + 1 < H) v11 = img[(yi + 1) * W + xi + 1];
    return v00 * (1 - wx) * (1 - wy) + v10 * wx * (1 - wy) +
           v01 * (1 - wx) * wy + v11 * wx * wy;
}

__global__ void k_corr_lookup(const float* __restrict__ c1,
                              const float* __restrict__ p0, const float* __restrict__ p1,
                              const float* __restrict__ p2, const float* __restrict__ p3,
                              float* __restrict__ cf, int Nb, int Hc, int Wc) {
    int t = blockIdx.x * blockDim.x + threadIdx.x;
    int P = Hc * Wc;
    int total = Nb * 324 * P;
    if (t >= total) return;
    int n  = t / (324 * P);
    int r  = t - n * (324 * P);
    int oc = r / P;
    int p  = r - oc * P;
    int lvl = oc / 81;
    int a3  = oc - lvl * 81;
    float dx = (float)(a3 / 9) - 4.f;
    float dy = (float)(a3 % 9) - 4.f;
    const float* base = (lvl == 0) ? p0 : (lvl == 1) ? p1 : (lvl == 2) ? p2 : p3;
    int Hs = Hc >> lvl, Ws = Wc >> lvl;
    const float* img = base + ((size_t)n * P + p) * Hs * Ws;
    float scale = 1.f / (float)(1 << lvl);
    float cx = c1[((size_t)n * 2 + 0) * P + p] * scale + dx;
    float cy = c1[((size_t)n * 2 + 1) * P + p] * scale + dy;
    cf[t] = bilin(img, Hs, Ws, cx, cy);
}

__global__ void k_split_cnet(const float* __restrict__ cnet, float* __restrict__ net,
                             float* __restrict__ inp, int Nb, int P) {
    int t = blockIdx.x * blockDim.x + threadIdx.x;
    int total = Nb * 128 * P;
    if (t >= total) return;
    int n = t / (128 * P);
    int r = t - n * (128 * P);
    int c = r / P, p = r - c * P;
    net[t] = tanhf(cnet[(((size_t)n * 256) + c) * P + p]);
    float v = cnet[(((size_t)n * 256) + 128 + c) * P + p];
    inp[t] = fmaxf(v, 0.f);
}

__global__ void k_upsample(const float* __restrict__ c1, const float* __restrict__ mask,
                           float* __restrict__ out, int Nb, int Hc, int Wc) {
    int t = blockIdx.x * blockDim.x + threadIdx.x;
    int P = Hc * Wc;
    int total = Nb * P * 16;
    if (t >= total) return;
    int ij = t & 15;
    int rest = t >> 4;
    int p = rest % P;
    int n = rest / P;
    int i = ij >> 2, j = ij & 3;
    int h = p / Wc, w = p - h * Wc;

    float lg[9];
    float mx = -3.0e38f;
    #pragma unroll
    for (int k = 0; k < 9; ++k) {
        lg[k] = mask[((size_t)n * 144 + k * 16 + i * 4 + j) * P + p];
        mx = fmaxf(mx, lg[k]);
    }
    float s = 0.f;
    #pragma unroll
    for (int k = 0; k < 9; ++k) { lg[k] = expf(lg[k] - mx); s += lg[k]; }
    float inv = 1.f / s;

    int HO = Hc * 4, WO = Wc * 4;
    #pragma unroll
    for (int c = 0; c < 2; ++c) {
        float acc = 0.f;
        #pragma unroll
        for (int k = 0; k < 9; ++k) {
            int di = k / 3, dj = k - di * 3;
            int hh = h + di - 1, ww = w + dj - 1;
            if (hh >= 0 && hh < Hc && ww >= 0 && ww < Wc) {
                float f = c1[((size_t)n * 2 + c) * P + hh * Wc + ww] -
                          (c == 0 ? (float)ww : (float)hh);
                acc += lg[k] * 4.f * f;
            }
        }
        acc *= inv;
        out[(((size_t)n * 2 + c) * HO + (h * 4 + i)) * WO + (w * 4 + j)] = acc;
    }
}

// ---------------------------------------------------------------------------
extern "C" void kernel_launch(void* const* d_in, const int* in_sizes, int n_in,
                              void* d_out, int out_size, void* d_ws, size_t ws_size,
                              hipStream_t stream) {
    (void)in_sizes; (void)n_in; (void)out_size; (void)ws_size;

    const int NB = 4, HIMG = 160, HC = 40, WC = 40, P = HC * WC, PIMG = HIMG * HIMG;
    const int ITERS = 6;

    auto IN = [&](int i) { return (const float*)d_in[i]; };
    // param indices (setup_inputs dict order):
    // 0 img1, 1 img2, 2 ctx, 3..10 fnet, 11..18 cnet, 19..24 resnet,
    // 25 qw 26 qb 27 kw 28 kb 29 vw 30 vb 31 gamma,
    // 32..41 motion enc, 42..47 h-GRU, 48..53 v-GRU, 54..61 heads.

    size_t off = 0;
    auto A = [&](size_t nfl) { float* pp = (float*)d_ws + off; off += nfl; return pp; };

    float* img1n = A((size_t)NB * PIMG);
    float* img2n = A((size_t)NB * PIMG);
    float* ctxn  = A((size_t)NB * 3 * PIMG);
    float* e1    = A((size_t)NB * 64 * 6400);   // also reused as `head`
    float* e2    = A((size_t)NB * 96 * P);
    float* e3    = A((size_t)NB * 128 * P);
    float* fm1   = A((size_t)NB * 256 * P);
    float* fm2   = A((size_t)NB * 256 * P);
    float* corr  = A((size_t)NB * P * P);
    float* pyr1  = A((size_t)NB * P * 400);
    float* pyr2  = A((size_t)NB * P * 100);
    float* pyr3  = A((size_t)NB * P * 25);
    float* rb1   = A((size_t)NB * 32 * PIMG);   // reused as hx  (== NB*512*P)
    float* rb2   = A((size_t)NB * 32 * PIMG);   // reused as qin (== NB*512*P)
    float* r3o   = A((size_t)NB * 3 * PIMG);
    float* cu    = A((size_t)NB * 3 * PIMG);
    float* cnet  = A((size_t)NB * 256 * P);
    float* net   = A((size_t)NB * 128 * P);
    float* inp   = A((size_t)NB * 128 * P);
    float* qb    = A((size_t)NB * 128 * P);
    float* kb2   = A((size_t)NB * 128 * P);
    float* attn  = A((size_t)NB * P * P);
    float* c1    = A((size_t)NB * 2 * P);
    float* flow  = A((size_t)NB * 2 * P);
    float* cf    = A((size_t)NB * 324 * P);
    float* mc1o  = A((size_t)NB * 256 * P);
    float* catcf = A((size_t)NB * 256 * P);
    float* mf1o  = A((size_t)NB * 128 * P);
    float* motion= A((size_t)NB * 128 * P);
    float* vb    = A((size_t)NB * 128 * P);
    float* agg   = A((size_t)NB * 128 * P);
    float* mg    = A((size_t)NB * 128 * P);
    float* xb    = A((size_t)NB * 384 * P);
    float* zb    = A((size_t)NB * 128 * P);
    float* rg    = A((size_t)NB * 128 * P);
    float* qg    = A((size_t)NB * 128 * P);
    float* dfl   = A((size_t)NB * 2 * P);
    float* maskb = A((size_t)NB * 144 * P);
    float* pmin  = A(256);
    float* pmax  = A(256);
    float* mm    = A(2);
    float* hx    = rb1;   // NB*512*P floats
    float* qin   = rb2;   // NB*512*P floats
    float* head  = e1;    // NB*256*P floats

    auto EW = [&](int n) { return dim3((n + 255) / 256); };

    auto conv = [&](const float* x, int wi, int bi, float* y,
                    int Nb, int Cin, int Hi, int Wi, int Cout, int Ho, int Wo,
                    int KH, int KW, int s, int pt, int pl,
                    int c0, int Ctot, float alpha, int act) {
        dim3 g((Nb * Ho * Wo + 15) / 16, (Cout + 63) / 64);
        k_wmma_conv<<<g, dim3(128), 0, stream>>>(x, IN(wi), IN(bi), y,
            Nb, Cin, Hi, Wi, Cout, Ho, Wo, KH, KW, s, pt, pl, c0, Ctot, alpha, act);
    };

    auto gemm = [&](const float* a, const float* b, float* o,
                    int M, int Ncols, int K, int tA, int tB, float alpha) {
        dim3 g((Ncols + 15) / 16, (M + 63) / 64, NB);
        k_wmma_gemm<<<g, dim3(128), 0, stream>>>(a, b, o, M, Ncols, K, tA, tB, alpha);
    };

    auto norm = [&](float* buf, int C, int Pn, bool inst) {
        if (inst) k_inorm<<<dim3(NB * C), dim3(256), 0, stream>>>(buf, Pn, 1);
        else      k_bnorm<<<dim3(C), dim3(256), 0, stream>>>(buf, NB, C, Pn, 1);
    };

    auto encoder = [&](const float* x, int Cin, int base, bool inst, float* fmap) {
        conv(x, base + 0, base + 1, e1, NB, Cin, 160, 160, 64, 80, 80, 7, 7, 2, 2, 2, 0, 64, 1.f, 0);
        norm(e1, 64, 6400, inst);
        conv(e1, base + 2, base + 3, e2, NB, 64, 80, 80, 96, 40, 40, 3, 3, 2, 0, 0, 0, 96, 1.f, 0);
        norm(e2, 96, P, inst);
        conv(e2, base + 4, base + 5, e3, NB, 96, 40, 40, 128, 40, 40, 3, 3, 1, 1, 1, 0, 128, 1.f, 0);
        norm(e3, 128, P, inst);
        conv(e3, base + 6, base + 7, fmap, NB, 128, 40, 40, 256, 40, 40, 1, 1, 1, 0, 0, 0, 256, 1.f, 0);
    };

    // ---- input normalization ----
    k_scale<<<EW(NB * PIMG), dim3(256), 0, stream>>>(IN(0), img1n, NB * PIMG);
    k_scale<<<EW(NB * PIMG), dim3(256), 0, stream>>>(IN(1), img2n, NB * PIMG);
    k_scale<<<EW(NB * 3 * PIMG), dim3(256), 0, stream>>>(IN(2), ctxn, NB * 3 * PIMG);

    // ---- feature encoders (instance norm) ----
    encoder(img1n, 1, 3, true, fm1);
    encoder(img2n, 1, 3, true, fm2);

    // ---- correlation volume + pyramid ----
    gemm(fm1, fm2, corr, P, P, 256, 1, 0, 1.f / 16.f);
    k_avgpool<<<EW(NB * P * 400), dim3(256), 0, stream>>>(corr, pyr1, NB * P, 20, 20);
    k_avgpool<<<EW(NB * P * 100), dim3(256), 0, stream>>>(pyr1, pyr2, NB * P, 10, 10);
    k_avgpool<<<EW(NB * P * 25), dim3(256), 0, stream>>>(pyr2, pyr3, NB * P, 5, 5);

    // ---- resnet refinement of context + min/max renorm ----
    conv(ctxn, 19, 20, rb1, NB, 3, 160, 160, 32, 160, 160, 3, 3, 1, 1, 1, 0, 32, 1.f, 1);
    conv(rb1, 21, 22, rb2, NB, 32, 160, 160, 32, 160, 160, 3, 3, 1, 1, 1, 0, 32, 1.f, 1);
    conv(rb2, 23, 24, r3o, NB, 32, 160, 160, 3, 160, 160, 3, 3, 1, 1, 1, 0, 3, 1.f, 0);
    k_resadd<<<EW(NB * 3 * PIMG), dim3(256), 0, stream>>>(cu, ctxn, r3o, NB * 3 * PIMG);
    k_minmax_part<<<dim3(256), dim3(256), 0, stream>>>(cu, NB * 3 * PIMG, pmin, pmax);
    k_minmax_fin<<<dim3(1), dim3(256), 0, stream>>>(pmin, pmax, mm, 256);
    k_renorm<<<EW(NB * 3 * PIMG), dim3(256), 0, stream>>>(
        cu, mm, (float*)d_out + (size_t)ITERS * NB * 2 * PIMG, NB * 3 * PIMG);

    // ---- context encoder (batch norm), split into net / inp ----
    encoder(cu, 3, 11, false, cnet);
    k_split_cnet<<<EW(NB * 128 * P), dim3(256), 0, stream>>>(cnet, net, inp, NB, P);

    // ---- GMA attention ----
    conv(inp, 25, 26, qb, NB, 128, 40, 40, 128, 40, 40, 1, 1, 1, 0, 0, 0, 128, 1.f, 0);
    conv(inp, 27, 28, kb2, NB, 128, 40, 40, 128, 40, 40, 1, 1, 1, 0, 0, 0, 128, 1.f, 0);
    gemm(qb, kb2, attn, P, P, 128, 1, 0, 0.0883883476f);
    k_softmax_rows<<<dim3(NB * P), dim3(256), 0, stream>>>(attn, P);

    // ---- iterative refinement ----
    k_coords_init<<<EW(NB * P), dim3(256), 0, stream>>>(c1, NB, HC, WC);

    for (int it = 0; it < ITERS; ++it) {
        k_flow<<<EW(NB * P), dim3(256), 0, stream>>>(c1, flow, motion, NB, HC, WC);
        k_corr_lookup<<<EW(NB * 324 * P), dim3(256), 0, stream>>>(
            c1, corr, pyr1, pyr2, pyr3, cf, NB, HC, WC);

        // motion encoder
        conv(cf, 32, 33, mc1o, NB, 324, 40, 40, 256, 40, 40, 1, 1, 1, 0, 0, 0, 256, 1.f, 1);
        conv(mc1o, 34, 35, catcf, NB, 256, 40, 40, 192, 40, 40, 3, 3, 1, 1, 1, 0, 256, 1.f, 1);
        conv(flow, 36, 37, mf1o, NB, 2, 40, 40, 128, 40, 40, 7, 7, 1, 3, 3, 0, 128, 1.f, 1);
        conv(mf1o, 38, 39, catcf, NB, 128, 40, 40, 64, 40, 40, 3, 3, 1, 1, 1, 192, 256, 1.f, 1);
        conv(catcf, 40, 41, motion, NB, 256, 40, 40, 126, 40, 40, 3, 3, 1, 1, 1, 0, 128, 1.f, 1);

        // global motion aggregation: agg = attn x v
        conv(motion, 29, 30, vb, NB, 128, 40, 40, 128, 40, 40, 1, 1, 1, 0, 0, 0, 128, 1.f, 0);
        gemm(vb, attn, agg, 128, P, P, 0, 1, 1.f);
        k_axpy_gamma<<<EW(NB * 128 * P), dim3(256), 0, stream>>>(mg, motion, agg, IN(31), NB * 128 * P);

        // x = [inp, motion, motion_g]
        k_copy_ch<<<EW(NB * 128 * P), dim3(256), 0, stream>>>(inp, xb, NB, 128, 384, 0, P);
        k_copy_ch<<<EW(NB * 128 * P), dim3(256), 0, stream>>>(motion, xb, NB, 128, 384, 128, P);
        k_copy_ch<<<EW(NB * 128 * P), dim3(256), 0, stream>>>(mg, xb, NB, 128, 384, 256, P);

        // --- GRU horizontal (1x5) ---
        k_copy_ch<<<EW(NB * 128 * P), dim3(256), 0, stream>>>(net, hx, NB, 128, 512, 0, P);
        k_copy_ch<<<EW(NB * 384 * P), dim3(256), 0, stream>>>(xb, hx, NB, 384, 512, 128, P);
        conv(hx, 42, 43, zb, NB, 512, 40, 40, 128, 40, 40, 1, 5, 1, 0, 2, 0, 128, 1.f, 2);
        conv(hx, 44, 45, rg, NB, 512, 40, 40, 128, 40, 40, 1, 5, 1, 0, 2, 0, 128, 1.f, 2);
        k_mul_into<<<EW(NB * 128 * P), dim3(256), 0, stream>>>(rg, net, qin, NB, 128, 512, 0, P);
        k_copy_ch<<<EW(NB * 384 * P), dim3(256), 0, stream>>>(xb, qin, NB, 384, 512, 128, P);
        conv(qin, 46, 47, qg, NB, 512, 40, 40, 128, 40, 40, 1, 5, 1, 0, 2, 0, 128, 1.f, 3);
        k_gru_update<<<EW(NB * 128 * P), dim3(256), 0, stream>>>(net, zb, qg, NB * 128 * P);

        // --- GRU vertical (5x1) ---
        k_copy_ch<<<EW(NB * 128 * P), dim3(256), 0, stream>>>(net, hx, NB, 128, 512, 0, P);
        conv(hx, 48, 49, zb, NB, 512, 40, 40, 128, 40, 40, 5, 1, 1, 2, 0, 0, 128, 1.f, 2);
        conv(hx, 50, 51, rg, NB, 512, 40, 40, 128, 40, 40, 5, 1, 1, 2, 0, 0, 128, 1.f, 2);
        k_mul_into<<<EW(NB * 128 * P), dim3(256), 0, stream>>>(rg, net, qin, NB, 128, 512, 0, P);
        conv(qin, 52, 53, qg, NB, 512, 40, 40, 128, 40, 40, 5, 1, 1, 2, 0, 0, 128, 1.f, 3);
        k_gru_update<<<EW(NB * 128 * P), dim3(256), 0, stream>>>(net, zb, qg, NB * 128 * P);

        // --- flow / mask heads ---
        conv(net, 54, 55, head, NB, 128, 40, 40, 256, 40, 40, 3, 3, 1, 1, 1, 0, 256, 1.f, 1);
        conv(head, 56, 57, dfl, NB, 256, 40, 40, 2, 40, 40, 3, 3, 1, 1, 1, 0, 2, 1.f, 0);
        conv(net, 58, 59, head, NB, 128, 40, 40, 256, 40, 40, 3, 3, 1, 1, 1, 0, 256, 1.f, 1);
        conv(head, 60, 61, maskb, NB, 256, 40, 40, 144, 40, 40, 1, 1, 1, 0, 0, 0, 144, 0.25f, 0);

        k_add_inplace<<<EW(NB * 2 * P), dim3(256), 0, stream>>>(c1, dfl, NB * 2 * P);
        k_upsample<<<EW(NB * P * 16), dim3(256), 0, stream>>>(
            c1, maskb, (float*)d_out + (size_t)it * NB * 2 * PIMG, NB, HC, WC);
    }
}
